// VectorQuantizer_25486335935226
// MI455X (gfx1250) — matrix operand order repaired and checked by reading, not compiled, not run on hardware
//
#include <hip/hip_runtime.h>

typedef __attribute__((ext_vector_type(16))) _Float16 v16h;
typedef __attribute__((ext_vector_type(8)))  _Float16 v8h;
typedef __attribute__((ext_vector_type(8)))  float    v8f;
typedef __attribute__((ext_vector_type(4)))  float    v4f;

#define NUM_E   8192
#define DIM     256
#define N_VEC   65536
#define QELEMS  16777216   // 65536 * 256
#define FLT_BIG 3.4028235e38f

// ---------------------------------------------------------------------------
// Kernel 1: convert embeddings f32 -> f16 (for WMMA) and compute exact f32
// squared norms. One wave (32 lanes) per code; each lane handles 8 dims.
// ---------------------------------------------------------------------------
__global__ __launch_bounds__(256)
void vq_prep(const float* __restrict__ emb, _Float16* __restrict__ e16,
             float* __restrict__ enorm) {
    const int wave = threadIdx.x >> 5;
    const int lane = threadIdx.x & 31;
    const int code = blockIdx.x * 8 + wave;

    const float* src = emb + (size_t)code * DIM + lane * 8;
    v4f a = *(const v4f*)src;
    v4f b = *(const v4f*)(src + 4);

    v8h h;
#pragma unroll
    for (int i = 0; i < 4; ++i) {
        h[i]     = (_Float16)a[i];
        h[4 + i] = (_Float16)b[i];
    }
    *(v8h*)(e16 + (size_t)code * DIM + lane * 8) = h;

    float s = 0.f;
#pragma unroll
    for (int i = 0; i < 4; ++i) s += a[i] * a[i] + b[i] * b[i];
#pragma unroll
    for (int m = 16; m >= 1; m >>= 1) s += __shfl_xor(s, m, 32);
    if (lane == 0) enorm[code] = s;
}

// ---------------------------------------------------------------------------
// Kernel 2: WMMA nearest-code search.
// Block = 256 threads = 8 waves. Each wave owns 32 rows of x (two 16-row A
// tiles kept in VGPRs for the whole K sweep). For each 16-code tile:
//   8 B fragments, 16 v_wmma_f32_16x16x32_f16 (two independent C chains),
//   then per-lane running min of (||e||^2 - 2*dot).
// C-matrix layout (per ISA): lane = 16*half + col holds column N=col,
// VGPR r holds row M = r + 8*half.
// ---------------------------------------------------------------------------
__global__ __launch_bounds__(256)
void vq_argmin(const float* __restrict__ x, const _Float16* __restrict__ e16,
               const float* __restrict__ enorm, int* __restrict__ idx_out) {
    const int lane = threadIdx.x & 31;
    const int wave = threadIdx.x >> 5;
    const int col  = lane & 15;
    const int half = lane >> 4;

    const int rowBase0 = blockIdx.x * 256 + wave * 32;  // first 16-row tile
    const int rowBase1 = rowBase0 + 16;                 // second 16-row tile

    // ---- Load + convert A fragments (x rows) once; k order used here (contiguous, same as B):
    // lanes<16 hold K = 32f+[0..15];
    // lanes>=16 hold K = 32f+16+[0..15].
    v16h a0[8], a1[8];
    {
        const float* r0 = x + (size_t)(rowBase0 + col) * DIM;
        const float* r1 = x + (size_t)(rowBase1 + col) * DIM;
#pragma unroll
        for (int f = 0; f < 8; ++f) {
            const int k0 = f * 32 + half * 16;
            const int k1 = k0 + 8;
            v4f pa = *(const v4f*)(r0 + k0);
            v4f pb = *(const v4f*)(r0 + k0 + 4);
            v4f pc = *(const v4f*)(r0 + k1);
            v4f pd = *(const v4f*)(r0 + k1 + 4);
            v16h t;
#pragma unroll
            for (int i = 0; i < 4; ++i) {
                t[i]      = (_Float16)pa[i];
                t[4 + i]  = (_Float16)pb[i];
                t[8 + i]  = (_Float16)pc[i];
                t[12 + i] = (_Float16)pd[i];
            }
            a0[f] = t;

            v4f qa = *(const v4f*)(r1 + k0);
            v4f qb = *(const v4f*)(r1 + k0 + 4);
            v4f qc = *(const v4f*)(r1 + k1);
            v4f qd = *(const v4f*)(r1 + k1 + 4);
            v16h u;
#pragma unroll
            for (int i = 0; i < 4; ++i) {
                u[i]      = (_Float16)qa[i];
                u[4 + i]  = (_Float16)qb[i];
                u[8 + i]  = (_Float16)qc[i];
                u[12 + i] = (_Float16)qd[i];
            }
            a1[f] = u;
        }
    }

    float minv[16];
    int   mini[16];
#pragma unroll
    for (int r = 0; r < 16; ++r) { minv[r] = FLT_BIG; mini[r] = 0; }

    // ---- Stream all codes, 16 per tile.
    for (int jt = 0; jt < NUM_E / 16; ++jt) {
        const int j = jt * 16 + col;          // this lane's code column
        const float en = enorm[j];

        v8f c0 = {};
        v8f c1 = {};
#pragma unroll
        for (int f = 0; f < 8; ++f) {
            // B in the same contiguous k order as A: lane holds 16 consecutive K values for
            // column N=col, K base = 32f + 16*half. 32B contiguous per lane.
            const _Float16* bp = e16 + (size_t)j * DIM + f * 32 + half * 16;
            v16h b = *(const v16h*)bp;
            c0 = __builtin_amdgcn_wmma_f32_16x16x32_f16(
                false, a0[f], false, b, (short)0, c0, false, false);
            c1 = __builtin_amdgcn_wmma_f32_16x16x32_f16(
                false, a1[f], false, b, (short)0, c1, false, false);
        }

#pragma unroll
        for (int r = 0; r < 8; ++r) {
            const float d0 = en - 2.0f * c0[r];
            if (d0 < minv[r])     { minv[r]     = d0; mini[r]     = j; }
            const float d1 = en - 2.0f * c1[r];
            if (d1 < minv[8 + r]) { minv[8 + r] = d1; mini[8 + r] = j; }
        }
    }

    // ---- Reduce across the 16 lanes of each half (columns of the row),
    // tie-breaking toward the lowest index (matches jnp.argmin).
#pragma unroll
    for (int r = 0; r < 16; ++r) {
        float v = minv[r];
        int   i = mini[r];
#pragma unroll
        for (int m = 8; m >= 1; m >>= 1) {
            const float ov = __shfl_xor(v, m, 32);
            const int   oi = __shfl_xor(i, m, 32);
            if (ov < v || (ov == v && oi < i)) { v = ov; i = oi; }
        }
        if (col == 0) {
            const int tileBase = (r < 8) ? rowBase0 : rowBase1;
            const int row = tileBase + (r & 7) + half * 8;
            idx_out[row] = i;
        }
    }
}

// ---------------------------------------------------------------------------
// Kernel 3: gather quantized rows (full f32 embeddings) + per-block partial
// of sum((q - x)^2). One wave per row; deterministic block partials.
// ---------------------------------------------------------------------------
__global__ __launch_bounds__(256)
void vq_gather(const float* __restrict__ x, const float* __restrict__ emb,
               const int* __restrict__ idx, float* __restrict__ qout,
               float* __restrict__ partials) {
    const int wave = threadIdx.x >> 5;
    const int lane = threadIdx.x & 31;
    const int row  = blockIdx.x * 8 + wave;
    const int code = idx[row];

    const float* e  = emb + (size_t)code * DIM + lane * 8;
    const float* xs = x   + (size_t)row  * DIM + lane * 8;
    float* qo = qout + (size_t)row * DIM + lane * 8;

    v4f e0 = *(const v4f*)e;
    v4f e1 = *(const v4f*)(e + 4);
    v4f x0 = *(const v4f*)xs;
    v4f x1 = *(const v4f*)(xs + 4);

    *(v4f*)qo       = e0;
    *(v4f*)(qo + 4) = e1;

    float s = 0.f;
#pragma unroll
    for (int i = 0; i < 4; ++i) {
        const float d0 = e0[i] - x0[i];
        const float d1 = e1[i] - x1[i];
        s += d0 * d0 + d1 * d1;
    }
#pragma unroll
    for (int m = 16; m >= 1; m >>= 1) s += __shfl_xor(s, m, 32);

    __shared__ float wsum[8];
    if (lane == 0) wsum[wave] = s;
    __syncthreads();
    if (threadIdx.x == 0) {
        float t = 0.f;
#pragma unroll
        for (int w = 0; w < 8; ++w) t += wsum[w];
        partials[blockIdx.x] = t;
    }
}

// ---------------------------------------------------------------------------
// Kernel 4: deterministic fixed-order final reduction of 8192 partials.
// loss = 1.25 * mean((q - x)^2)   (q_latent + COMMITMENT_COST * e_latent,
// identical values in eval mode).
// ---------------------------------------------------------------------------
__global__ __launch_bounds__(256)
void vq_loss_final(const float* __restrict__ partials, float* __restrict__ loss) {
    __shared__ float sm[256];
    float s = 0.f;
    for (int i = threadIdx.x; i < 8192; i += 256) s += partials[i];
    sm[threadIdx.x] = s;
    __syncthreads();
    for (int off = 128; off >= 1; off >>= 1) {
        if ((int)threadIdx.x < off) sm[threadIdx.x] += sm[threadIdx.x + off];
        __syncthreads();
    }
    if (threadIdx.x == 0) *loss = sm[0] * (1.25f / (float)QELEMS);
}

// ---------------------------------------------------------------------------
extern "C" void kernel_launch(void* const* d_in, const int* in_sizes, int n_in,
                              void* d_out, int out_size, void* d_ws, size_t ws_size,
                              hipStream_t stream) {
    const float* x   = (const float*)d_in[0];   // [16,64,64,256] f32
    const float* emb = (const float*)d_in[1];   // [8192,256] f32

    float* out      = (float*)d_out;
    float* qout     = out;                       // QELEMS floats
    float* loss     = out + QELEMS;              // 1 float
    int*   idx_out  = (int*)(out + QELEMS + 1);  // 65536 int32

    // workspace layout
    _Float16* e16     = (_Float16*)d_ws;                                   // 4 MB
    float*    enorm   = (float*)((char*)d_ws + (size_t)NUM_E * DIM * 2);   // 32 KB
    float*    partial = enorm + NUM_E;                                     // 32 KB

    vq_prep    <<<NUM_E / 8,  256, 0, stream>>>(emb, e16, enorm);
    vq_argmin  <<<N_VEC / 256, 256, 0, stream>>>(x, e16, enorm, idx_out);
    vq_gather  <<<N_VEC / 8,  256, 0, stream>>>(x, emb, idx_out, qout, partial);
    vq_loss_final<<<1,        256, 0, stream>>>(partial, loss);
}
